// GAT_79061757985147
// MI455X (gfx1250) — compile-verified, matrix operand order checked by reading, not run on hardware
//
#include <hip/hip_runtime.h>
#include <hip/hip_bf16.h>

#define N_NODES 50000
#define E_EDGES 800000
#define E_TOT   (E_EDGES + N_NODES)
#define GRAPHS  64
#define F_IN    128
#define HID     64
#define HEADS   2
#define FDIM    (HEADS * HID)   /* 128 */
#define OUT_DIM 10
#define NEG_SLOPE 0.2f

typedef __attribute__((ext_vector_type(8)))  float          v8f;
typedef __attribute__((ext_vector_type(16))) __bf16         v16bf;
typedef __attribute__((ext_vector_type(8)))  unsigned short ushort8;
typedef __attribute__((ext_vector_type(16))) unsigned short ushort16;

__device__ __forceinline__ unsigned short f32_to_bf16(float f) {
    unsigned int u = __float_as_uint(f);
    u = (u + 0x7FFFu + ((u >> 16) & 1u)) >> 16;   // round-to-nearest-even
    return (unsigned short)u;
}

__device__ __forceinline__ void atomicMaxF(float* addr, float value) {
    // branch-split float max; addr must be initialized to -0.0f (0x80000000)
    if (value >= 0.0f)
        atomicMax((int*)addr, __float_as_int(value));
    else
        atomicMin((unsigned int*)addr, __float_as_uint(value));
}

// ---------------------------------------------------------------- W -> bf16^T
__global__ void cvt_transpose_kernel(const float* __restrict__ W,
                                     unsigned short* __restrict__ Wt) {
    int idx = blockIdx.x * blockDim.x + threadIdx.x;  // 0 .. 128*128-1
    if (idx >= FDIM * FDIM) return;
    int col = idx >> 7;
    int k   = idx & 127;
    Wt[col * FDIM + k] = f32_to_bf16(W[k * FDIM + col]);
}

// ------------------------------------------------------- WMMA GEMM  D = A*W
// A: f32 [N,128], Wt: bf16 [128 cols][128 k] (pre-transposed), D: f32 [N,128]
// block = 256 threads = 8 waves; block computes rows [16*bx,16*bx+16), all 128 cols.
__global__ __launch_bounds__(256)
void gemm_bf16_kernel(const float* __restrict__ A,
                      const unsigned short* __restrict__ Wt,
                      float* __restrict__ Dst) {
    __shared__ __align__(16) unsigned short As[16 * FDIM];

    const int tid  = threadIdx.x;
    const int lane = tid & 31;
    const int wv   = tid >> 5;                // 0..7 -> 16-col tile
    const int row0 = blockIdx.x * 16;

    // stage A tile (16x128 f32) into LDS as bf16, 2 x float4 per thread
    const float4*  Ap   = (const float4*)(A + (size_t)row0 * FDIM);
    unsigned int*  As32 = (unsigned int*)As;
    #pragma unroll
    for (int j = 0; j < 2; ++j) {
        int v = tid * 2 + j;                  // 0..511 float4s
        float4 f = Ap[v];
        As32[v * 2 + 0] = (unsigned)f32_to_bf16(f.x) | ((unsigned)f32_to_bf16(f.y) << 16);
        As32[v * 2 + 1] = (unsigned)f32_to_bf16(f.z) | ((unsigned)f32_to_bf16(f.w) << 16);
    }

    // B fragment pointers (hot in L2/WGP$); hint prefetch of this wave's column
    const int col = wv * 16 + (lane & 15);
    const ushort8* brow = (const ushort8*)(Wt + (size_t)col * FDIM);
    __builtin_prefetch(brow, 0, 3);           // global_prefetch_b8

    __syncthreads();

    // ISA 16-bit A 16x32 layout: lanes 0-15 hold K {0..7,16..23}, lanes 16-31 hold K {8..15,24..31}
    const int m  = lane & 15;
    const int kb = (lane >> 4) << 3;          // 0 or 8
    const ushort8* arow = (const ushort8*)(As + m * FDIM);

    v8f acc = {0.f, 0.f, 0.f, 0.f, 0.f, 0.f, 0.f, 0.f};
    #pragma unroll
    for (int kk = 0; kk < 4; ++kk) {          // K = 128 in 4 x 32 steps
        const int i0 = (kk * 32 + kb) >> 3;
        const int i1 = (kk * 32 + kb + 16) >> 3;
        ushort8 al = arow[i0], ah = arow[i1];
        ushort8 bl = brow[i0], bh = brow[i1];
        union { ushort16 u; v16bf v; } a, b;
        #pragma unroll
        for (int i = 0; i < 8; ++i) {
            a.u[i] = al[i]; a.u[i + 8] = ah[i];
            b.u[i] = bl[i]; b.u[i + 8] = bh[i];
        }
        acc = __builtin_amdgcn_wmma_f32_16x16x32_bf16(
                  false, a.v, false, b.v, (short)0, acc, false, false);
    }

    // D layout: lanes 0-15 -> M = r, lanes 16-31 -> M = r+8; N = lane&15
    const int mBase = (lane >> 4) << 3;
    #pragma unroll
    for (int r = 0; r < 8; ++r)
        Dst[(size_t)(row0 + mBase + r) * FDIM + col] = acc[r];
}

// ---------------------------------------------------- attention coefficients
__global__ void att_coef_kernel(const float* __restrict__ h,
                                const float* __restrict__ attS,
                                const float* __restrict__ attD,
                                float* __restrict__ asrc,
                                float* __restrict__ adst) {
    int t = blockIdx.x * blockDim.x + threadIdx.x;
    if (t >= N_NODES * HEADS) return;
    int n = t >> 1, hh = t & 1;
    const float* hp = h + (size_t)n * FDIM + hh * HID;
    float s1 = 0.f, s2 = 0.f;
    #pragma unroll 8
    for (int c = 0; c < HID; ++c) {
        float v = hp[c];
        s1 += v * attS[hh * HID + c];
        s2 += v * attD[hh * HID + c];
    }
    asrc[t] = s1;
    adst[t] = s2;
}

__global__ void init_softmax_kernel(float* __restrict__ mmax,
                                    float* __restrict__ denom) {
    int t = blockIdx.x * blockDim.x + threadIdx.x;
    if (t >= N_NODES * HEADS) return;
    mmax[t]  = __uint_as_float(0x80000000u);   // -0.0f
    denom[t] = 0.f;
}

// ------------------------------------------------ edge alpha + segment max
__global__ void alpha_max_kernel(const int* __restrict__ ei,
                                 const float* __restrict__ asrc,
                                 const float* __restrict__ adst,
                                 float* __restrict__ alpha,
                                 float* __restrict__ mmax) {
    int e = blockIdx.x * blockDim.x + threadIdx.x;
    if (e >= E_TOT) return;
    int s, d;
    if (e < E_EDGES) { s = ei[e]; d = ei[E_EDGES + e]; }
    else             { s = e - E_EDGES; d = s; }
    #pragma unroll
    for (int hh = 0; hh < HEADS; ++hh) {
        float v = asrc[s * HEADS + hh] + adst[d * HEADS + hh];
        v = v > 0.f ? v : NEG_SLOPE * v;
        alpha[(size_t)e * HEADS + hh] = v;
        atomicMaxF(&mmax[d * HEADS + hh], v);
    }
}

// ------------------------------------------------ exp + segment sum
__global__ void exp_sum_kernel(const int* __restrict__ ei,
                               float* __restrict__ alpha,
                               const float* __restrict__ mmax,
                               float* __restrict__ denom) {
    int e = blockIdx.x * blockDim.x + threadIdx.x;
    if (e >= E_TOT) return;
    int d = (e < E_EDGES) ? ei[E_EDGES + e] : (e - E_EDGES);
    #pragma unroll
    for (int hh = 0; hh < HEADS; ++hh) {
        float a = __expf(alpha[(size_t)e * HEADS + hh] - mmax[d * HEADS + hh]);
        alpha[(size_t)e * HEADS + hh] = a;
        atomicAdd(&denom[d * HEADS + hh], a);
    }
}

// ------------------------------------------------ normalize + scatter-add
// 128 threads per edge (one per channel); 2 edges per 256-thread block
__global__ __launch_bounds__(256)
void aggregate_kernel(const int* __restrict__ ei,
                      const float* __restrict__ alpha,
                      const float* __restrict__ denom,
                      const float* __restrict__ h,
                      float* __restrict__ agg) {
    int t = blockIdx.x * 256 + threadIdx.x;
    int e = t >> 7;
    int c = t & 127;
    if (e >= E_TOT) return;
    int s, d;
    if (e < E_EDGES) { s = ei[e]; d = ei[E_EDGES + e]; }
    else             { s = e - E_EDGES; d = s; }
    int hh = c >> 6;
    float w = alpha[(size_t)e * HEADS + hh] / (denom[d * HEADS + hh] + 1e-16f);
    atomicAdd(&agg[(size_t)d * FDIM + c], w * h[(size_t)s * FDIM + c]);
}

__global__ void bias_relu_kernel(float* __restrict__ x,
                                 const float* __restrict__ b) {
    int t = blockIdx.x * blockDim.x + threadIdx.x;
    if (t >= N_NODES * FDIM) return;
    x[t] = fmaxf(x[t] + b[t & 127], 0.f);
}

// ------------------------------------------------ mean pool over graphs
__global__ void pool_kernel(const float* __restrict__ x,
                            const int* __restrict__ batch,
                            float* __restrict__ sums,
                            float* __restrict__ cnts) {
    int n = blockIdx.x;
    int c = threadIdx.x;
    int g = batch[n];
    atomicAdd(&sums[g * FDIM + c], x[(size_t)n * FDIM + c]);
    if (c == 0) atomicAdd(&cnts[g], 1.0f);
}

__global__ void final_kernel(const float* __restrict__ sums,
                             const float* __restrict__ cnts,
                             const float* __restrict__ Wl,
                             const float* __restrict__ bl,
                             float* __restrict__ out) {
    int t = blockIdx.x * blockDim.x + threadIdx.x;
    if (t >= GRAPHS * OUT_DIM) return;
    int g = t / OUT_DIM, o = t % OUT_DIM;
    float inv = 1.0f / fmaxf(cnts[g], 1.0f);
    float acc = bl[o];
    #pragma unroll 8
    for (int k = 0; k < FDIM; ++k)
        acc += sums[g * FDIM + k] * inv * Wl[k * OUT_DIM + o];
    out[t] = acc;
}

// ============================================================================
extern "C" void kernel_launch(void* const* d_in, const int* in_sizes, int n_in,
                              void* d_out, int out_size, void* d_ws, size_t ws_size,
                              hipStream_t stream) {
    const float* x     = (const float*)d_in[0];
    const int*   ei    = (const int*)d_in[1];
    const int*   batch = (const int*)d_in[2];
    const float* W1    = (const float*)d_in[3];
    const float* attS1 = (const float*)d_in[4];
    const float* attD1 = (const float*)d_in[5];
    const float* b1    = (const float*)d_in[6];
    const float* W2    = (const float*)d_in[7];
    const float* attS2 = (const float*)d_in[8];
    const float* attD2 = (const float*)d_in[9];
    const float* b2    = (const float*)d_in[10];
    const float* Wl    = (const float*)d_in[11];
    const float* bl    = (const float*)d_in[12];
    float* out = (float*)d_out;

    char* ws = (char*)d_ws;
    size_t off = 0;
    auto alloc = [&](size_t bytes) {
        void* p = ws + off;
        off = (off + bytes + 255) & ~(size_t)255;
        return p;
    };

    unsigned short* wt1  = (unsigned short*)alloc(FDIM * FDIM * 2);
    unsigned short* wt2  = (unsigned short*)alloc(FDIM * FDIM * 2);
    float* asrc  = (float*)alloc((size_t)N_NODES * HEADS * 4);
    float* adst  = (float*)alloc((size_t)N_NODES * HEADS * 4);
    float* mmax  = (float*)alloc((size_t)N_NODES * HEADS * 4);
    float* denom = (float*)alloc((size_t)N_NODES * HEADS * 4);
    float* alpha = (float*)alloc((size_t)E_TOT * HEADS * 4);
    float* hlin  = (float*)alloc((size_t)N_NODES * FDIM * 4);
    float* bufA  = (float*)alloc((size_t)N_NODES * FDIM * 4);
    float* bufB  = (float*)alloc((size_t)N_NODES * FDIM * 4);
    float* sums  = (float*)alloc((size_t)GRAPHS * FDIM * 4);
    float* cnts  = (float*)alloc((size_t)GRAPHS * 4);

    const int nh_blocks  = (N_NODES * HEADS + 255) / 256;
    const int e_blocks   = (E_TOT + 255) / 256;
    const int agg_blocks = (E_TOT + 1) / 2;            // 128 thr/edge, 256/block
    const int nf_blocks  = (N_NODES * FDIM + 255) / 256;

    auto run_layer = [&](const float* xin, const unsigned short* wt,
                         const float* attS, const float* attD, const float* bias,
                         float* agg) {
        gemm_bf16_kernel<<<N_NODES / 16, 256, 0, stream>>>(xin, wt, hlin);
        att_coef_kernel<<<nh_blocks, 256, 0, stream>>>(hlin, attS, attD, asrc, adst);
        init_softmax_kernel<<<nh_blocks, 256, 0, stream>>>(mmax, denom);
        hipMemsetAsync(agg, 0, (size_t)N_NODES * FDIM * 4, stream);
        alpha_max_kernel<<<e_blocks, 256, 0, stream>>>(ei, asrc, adst, alpha, mmax);
        exp_sum_kernel<<<e_blocks, 256, 0, stream>>>(ei, alpha, mmax, denom);
        aggregate_kernel<<<agg_blocks, 256, 0, stream>>>(ei, alpha, denom, hlin, agg);
        bias_relu_kernel<<<nf_blocks, 256, 0, stream>>>(agg, bias);
    };

    cvt_transpose_kernel<<<64, 256, 0, stream>>>(W1, wt1);
    cvt_transpose_kernel<<<64, 256, 0, stream>>>(W2, wt2);

    run_layer(x,    wt1, attS1, attD1, b1, bufA);
    run_layer(bufA, wt2, attS2, attD2, b2, bufB);

    hipMemsetAsync(sums, 0, (size_t)GRAPHS * FDIM * 4, stream);
    hipMemsetAsync(cnts, 0, (size_t)GRAPHS * 4, stream);
    pool_kernel<<<N_NODES, 128, 0, stream>>>(bufB, batch, sums, cnts);
    final_kernel<<<(GRAPHS * OUT_DIM + 63) / 64, 64, 0, stream>>>(sums, cnts, Wl, bl, out);
}
